// FusedKQnA_14328010900121
// MI455X (gfx1250) — matrix operand order repaired and checked by reading, not compiled, not run on hardware
//
#include <hip/hip_runtime.h>
#include <math.h>

// ---------------- CDNA5 WMMA / TDM types ----------------
typedef __attribute__((ext_vector_type(16))) __bf16    v16bf;
typedef __attribute__((ext_vector_type(8)))  float     v8f;
typedef __attribute__((ext_vector_type(4)))  unsigned  u32x4;
typedef __attribute__((ext_vector_type(4)))  int       i32x4;
typedef __attribute__((ext_vector_type(8)))  int       i32x8;

union Frag {            // 32 bytes = one wave32 bf16 A/B fragment per lane
    u32x4 u[2];
    v16bf v;
};

#if defined(__has_builtin)
#if __has_builtin(__builtin_amdgcn_tensor_load_to_lds) && __has_builtin(__builtin_amdgcn_s_wait_tensorcnt)
#define HAVE_TDM 1
#endif
#endif
#ifndef HAVE_TDM
#define HAVE_TDM 0
#endif

// dims
#define BD   32
#define HH   64
#define WW   64
#define CD   256
#define FEAT 256
#define HEADS 8
#define NQ   2
#define HDIM 32
#define NQH  16      // NQ*HEADS
#define HO   32
#define WO   32
#define M1   (BD*HH*WW)   // 131072 pixels
#define M3   (BD*HO*WO)   // 32768 pixels
#define NCOL1 272         // 256 (Wv) + 16 (qWk)

// ---------------- bf16 helpers (RNE) ----------------
__device__ __forceinline__ unsigned short f2bf(float f) {
    union { float f; unsigned u; } c; c.f = f;
    unsigned u = c.u;
    u += 0x7FFFu + ((u >> 16) & 1u);
    return (unsigned short)(u >> 16);
}
__device__ __forceinline__ float bf2f(unsigned short s) {
    union { unsigned u; float f; } c; c.u = ((unsigned)s) << 16;
    return c.f;
}
// pack (k,n) -> fragment-major offset for a 16-wide N tile
// B fragment k order used here (contiguous, same as A; 32x16 bf16): VGPR v, lane n<16: K=2v,2v+1, N=n ; lane>=16: K=16+2v..., N=n-16
__device__ __forceinline__ size_t b_frag_off(int k, int n) {
    int ntile = n >> 4, nn = n & 15, kc = k >> 5, kk = k & 31;
    int lane = nn + ((kk >> 4) << 4);
    return ((size_t)(ntile * 8 + kc) * 32 + lane) * 16 + (kk & 15);
}

// ---------------- prep: qWk fold, weight split/swizzle, rpe tables ----------------
__global__ void __launch_bounds__(256)
qna_prep_kernel(const float* __restrict__ query, const float* __restrict__ Wk,
                const float* __restrict__ Wv,    const float* __restrict__ Wo,
                const float* __restrict__ rpe_bias, const float* __restrict__ attn_scale_w,
                unsigned short* __restrict__ W1hi, unsigned short* __restrict__ W1lo,
                unsigned short* __restrict__ W3hi, unsigned short* __restrict__ W3lo,
                float* __restrict__ rpe_exp, float* __restrict__ scale_k)
{
    __shared__ float s_q[NQH][HDIM];
    const int tid = threadIdx.x;

    if (tid < NQH) {                       // normalized, depth-scaled queries
        int q = tid >> 3, h = tid & 7;
        float tmp[HDIM]; float nrm = 0.f;
        #pragma unroll
        for (int d = 0; d < HDIM; ++d) { tmp[d] = query[q*FEAT + h*HDIM + d]; nrm += tmp[d]*tmp[d]; }
        float s = 1.0f / ((sqrtf(nrm) + 1e-6f) * sqrtf((float)HDIM));
        #pragma unroll
        for (int d = 0; d < HDIM; ++d) s_q[tid][d] = tmp[d] * s;
    }
    __syncthreads();

    if (tid < 9 * NQH) {                   // exp(rpe) and exp(rpe)*scale (max-shift cancels in ratio)
        float r = expf(rpe_bias[tid]);
        rpe_exp[tid] = r;
        scale_k[tid] = r * attn_scale_w[tid];
    }

    // pack combined [Wv | qWk] (256 x 272) -> bf16 hi/lo, WMMA-B fragment layout
    for (int i = tid; i < NCOL1 * CD; i += blockDim.x) {
        int n = i % NCOL1, k = i / NCOL1;
        float val;
        if (n < FEAT) {
            val = Wv[k*FEAT + n];
        } else {
            int qh = n - FEAT, h = qh & 7;
            float a = 0.f;
            #pragma unroll
            for (int d = 0; d < HDIM; ++d) a += s_q[qh][d] * Wk[k*FEAT + h*HDIM + d];
            val = a;
        }
        unsigned short hi = f2bf(val);
        unsigned short lo = f2bf(val - bf2f(hi));
        size_t off = b_frag_off(k, n);
        W1hi[off] = hi; W1lo[off] = lo;
    }
    // pack Wo (256 x 256)
    for (int i = tid; i < FEAT * FEAT; i += blockDim.x) {
        int n = i & 255, k = i >> 8;
        float val = Wo[k*FEAT + n];
        unsigned short hi = f2bf(val);
        unsigned short lo = f2bf(val - bf2f(hi));
        size_t off = b_frag_off(k, n);
        W3hi[off] = hi; W3lo[off] = lo;
    }
}

// ---------------- split-bf16 WMMA GEMM: [M x 256] * [256 x 16*n_main(+16)] ----------------
// out_cost != nullptr => an extra 17th tile computes exp(cost) into out_cost [M x 16]
__global__ void __launch_bounds__(256)
qna_gemm_kernel(const float* __restrict__ A,
                const unsigned short* __restrict__ Whi,
                const unsigned short* __restrict__ Wlo,
                int n_main, float* __restrict__ out_main, float* __restrict__ out_cost)
{
    __shared__ __align__(16) unsigned short sA_hi[16][264];  // 264 = 256 + 8 pad (bank-spread)
    __shared__ __align__(16) unsigned short sA_lo[16][264];

    const int tid = threadIdx.x;
    const int m0  = blockIdx.x * 16;

#if HAVE_TDM
    // --- Tensor Data Mover: DMA the 16x256 fp32 A-tile into LDS, then split to bf16 ---
    __shared__ __align__(16) float sA_f32[16][256];
    {
        if (tid < 32) {   // one wave issues the TDM op (TDM ignores EXEC; avoid 8x duplicates)
            unsigned long long ga = (unsigned long long)(const void*)(A + (size_t)m0 * CD);
            unsigned lds = (unsigned)(unsigned long long)(const void*)&sA_f32[0][0]; // LDS aperture keeps offset in addr[31:0]
            // D# group 0: count=1 | lds_addr | global_addr[56:0] | type=2
            u32x4 g0 = { 1u, lds,
                         (unsigned)(ga & 0xffffffffu),
                         (unsigned)((ga >> 32) & 0x01ffffffu) | (2u << 30) };
            // D# group 1: data_size=2 (4B); tensor_dim0=256; tensor_dim1=16;
            //             tile_dim0=256; tile_dim1=16; tensor_dim0_stride=256
            i32x8 g1 = { 0x00020000,          // mask=0, data_size=4B, no barrier/iter/pad
                         (int)(256u << 16),   // tensor_dim0[15:0] in [31:16]
                         (int)(16u  << 16),   // tensor_dim0[31:16]=0 | tensor_dim1[15:0]
                         (int)(256u << 16),   // tensor_dim1[31:16]=0 | tile_dim0
                         16,                  // tile_dim1 | tile_dim2=0
                         256,                 // tensor_dim0_stride[31:0]
                         0, 0 };              // stride0 hi / tensor_dim1_stride
            i32x4 gz = { 0, 0, 0, 0 };
#if __clang_major__ >= 23
            i32x8 gz8 = { 0, 0, 0, 0, 0, 0, 0, 0 };
            __builtin_amdgcn_tensor_load_to_lds(g0, g1, gz, gz, gz8, 0);
#else
            __builtin_amdgcn_tensor_load_to_lds(g0, g1, gz, gz, 0);
#endif
            __builtin_amdgcn_s_wait_tensorcnt(0);
        }
        __syncthreads();
        const int row = tid >> 4;
        const int c0  = (tid & 15) << 4;
        const float* src = &sA_f32[row][c0];
        #pragma unroll
        for (int j = 0; j < 16; j += 4) {
            float4 f = *(const float4*)(src + j);
            float v[4] = {f.x, f.y, f.z, f.w};
            #pragma unroll
            for (int t = 0; t < 4; ++t) {
                unsigned short hi = f2bf(v[t]);
                sA_hi[row][c0 + j + t] = hi;
                sA_lo[row][c0 + j + t] = f2bf(v[t] - bf2f(hi));
            }
        }
    }
#else
    // --- fallback: cooperative global->VGPR->LDS staging with fp32->bf16 split ---
    {
        const int row = tid >> 4;
        const int c0  = (tid & 15) << 4;
        const float* src = A + (size_t)(m0 + row) * CD + c0;
        #pragma unroll
        for (int j = 0; j < 16; j += 4) {
            float4 f = *(const float4*)(src + j);
            float v[4] = {f.x, f.y, f.z, f.w};
            #pragma unroll
            for (int t = 0; t < 4; ++t) {
                unsigned short hi = f2bf(v[t]);
                sA_hi[row][c0 + j + t] = hi;
                sA_lo[row][c0 + j + t] = f2bf(v[t] - bf2f(hi));
            }
        }
    }
#endif
    __syncthreads();

    const int wave  = tid >> 5;
    const int lane  = tid & 31;
    const int arow  = lane & 15;
    const int abase = (lane < 16) ? 0 : 16;  // A k order used here (matches B packing): lane half h holds K=16h..16h+15
    const int n_total = n_main + (out_cost ? 1 : 0);

    for (int t = wave; t < n_total; t += 8) {
        v8f acc = {0.f,0.f,0.f,0.f,0.f,0.f,0.f,0.f};
        const unsigned short* wh = Whi + ((size_t)t * 256 + lane) * 16;
        const unsigned short* wl = Wlo + ((size_t)t * 256 + lane) * 16;
        #pragma unroll
        for (int kc = 0; kc < 8; ++kc) {
            Frag ah, al, bh, bl;
            const unsigned short* pa = &sA_hi[arow][kc * 32 + abase];
            ah.u[0] = *(const u32x4*)(pa);
            ah.u[1] = *(const u32x4*)(pa + 8);
            const unsigned short* pl = &sA_lo[arow][kc * 32 + abase];
            al.u[0] = *(const u32x4*)(pl);
            al.u[1] = *(const u32x4*)(pl + 8);
            const u32x4* pb = (const u32x4*)(wh + (size_t)kc * 512);
            bh.u[0] = pb[0]; bh.u[1] = pb[1];
            const u32x4* pc = (const u32x4*)(wl + (size_t)kc * 512);
            bl.u[0] = pc[0]; bl.u[1] = pc[1];
            if (kc < 7) {  // -> global_prefetch_b8 on the streamed B fragments
                __builtin_prefetch(wh + (size_t)(kc + 1) * 512, 0, 3);
                __builtin_prefetch(wl + (size_t)(kc + 1) * 512, 0, 3);
            }
            // fp32 via 2-term bf16 split: hi*hi + hi*lo + lo*hi
            acc = __builtin_amdgcn_wmma_f32_16x16x32_bf16(false, ah.v, false, bh.v, (short)0, acc, false, false);
            acc = __builtin_amdgcn_wmma_f32_16x16x32_bf16(false, ah.v, false, bl.v, (short)0, acc, false, false);
            acc = __builtin_amdgcn_wmma_f32_16x16x32_bf16(false, al.v, false, bh.v, (short)0, acc, false, false);
        }
        // C/D layout: VGPR r -> M = r + 8*(lane>=16), N = lane%16
        const int nn = lane & 15;
        const int mh = (lane >> 4) << 3;
        if (t < n_main) {
            float* dst = out_main + (size_t)(m0 + mh) * (n_main * 16) + t * 16 + nn;
            #pragma unroll
            for (int r = 0; r < 8; ++r) dst[(size_t)r * (n_main * 16)] = acc[r];
        } else {
            float* dst = out_cost + (size_t)(m0 + mh) * NQH + nn;
            #pragma unroll
            for (int r = 0; r < 8; ++r) dst[(size_t)r * NQH] = expf(acc[r]);
        }
    }
}

// ---------------- 3x3 stride-2 windowed softmax-style reduction ----------------
__global__ void __launch_bounds__(256)
qna_conv_kernel(const float* __restrict__ cexp,     // [B,64,64,16]
                const float* __restrict__ vbuf,     // [B,64,64,256]
                const float* __restrict__ rpe_exp,  // [9][16]
                const float* __restrict__ scale_k,  // [9][16]
                float* __restrict__ mid)            // [B,32,32,256]
{
    __shared__ float s_c[9][NQH];
    __shared__ float s_den[NQH];
    __shared__ float s_rpe[9 * NQH];
    __shared__ float s_scl[9 * NQH];

    const int tid = threadIdx.x;
    const int p   = blockIdx.x;
    const int b   = p >> 10;
    const int io  = (p >> 5) & 31;
    const int jo  = p & 31;

    if (tid < 9 * NQH) { s_rpe[tid] = rpe_exp[tid]; s_scl[tid] = scale_k[tid]; }
    __syncthreads();

    if (tid < NQH) {
        float den = 0.f;
        #pragma unroll
        for (int tap = 0; tap < 9; ++tap) {
            int y = 2 * io - 1 + tap / 3;
            int x = 2 * jo - 1 + tap % 3;
            float c = 0.f;
            if ((unsigned)y < (unsigned)HH && (unsigned)x < (unsigned)WW)
                c = cexp[(((size_t)b * HH + y) * WW + x) * NQH + tid];
            s_c[tap][tid] = c;
            den += s_rpe[tap * NQH + tid] * c;
        }
        s_den[tid] = den;
    }
    __syncthreads();

    const int f = tid;          // feature = h*32 + d
    const int h = f >> 5;
    float a0 = 0.f, a1 = 0.f;
    #pragma unroll
    for (int tap = 0; tap < 9; ++tap) {
        int y = 2 * io - 1 + tap / 3;
        int x = 2 * jo - 1 + tap % 3;
        if ((unsigned)y < (unsigned)HH && (unsigned)x < (unsigned)WW) {
            float vv = vbuf[(((size_t)b * HH + y) * WW + x) * FEAT + f];
            a0 += s_scl[tap * NQH + h]     * s_c[tap][h]     * vv;   // q = 0
            a1 += s_scl[tap * NQH + 8 + h] * s_c[tap][8 + h] * vv;   // q = 1
        }
    }
    mid[(size_t)p * FEAT + f] = a0 / s_den[h] + a1 / s_den[8 + h];
}

// ---------------- launch ----------------
extern "C" void kernel_launch(void* const* d_in, const int* in_sizes, int n_in,
                              void* d_out, int out_size, void* d_ws, size_t ws_size,
                              hipStream_t stream) {
    const float* x            = (const float*)d_in[0];
    const float* query        = (const float*)d_in[1];
    const float* Wk           = (const float*)d_in[2];
    const float* Wv           = (const float*)d_in[3];
    const float* rpe_bias     = (const float*)d_in[4];
    const float* attn_scale_w = (const float*)d_in[5];
    const float* Wo           = (const float*)d_in[6];

    char* ws = (char*)d_ws;
    size_t off = 0;
    auto take = [&](size_t bytes) { char* p = ws + off; off = (off + bytes + 255) & ~(size_t)255; return p; };

    unsigned short* W1hi = (unsigned short*)take((size_t)NCOL1 * CD * 2);
    unsigned short* W1lo = (unsigned short*)take((size_t)NCOL1 * CD * 2);
    unsigned short* W3hi = (unsigned short*)take((size_t)FEAT * FEAT * 2);
    unsigned short* W3lo = (unsigned short*)take((size_t)FEAT * FEAT * 2);
    float* rpe  = (float*)take(9 * NQH * 4);
    float* scl  = (float*)take(9 * NQH * 4);
    float* cbuf = (float*)take((size_t)M1 * NQH * 4);     // exp(cost)  [131072 x 16]
    float* vbuf = (float*)take((size_t)M1 * FEAT * 4);    // values     [131072 x 256]
    float* mid  = (float*)take((size_t)M3 * FEAT * 4);    // pre-Wo     [32768 x 256]

    qna_prep_kernel<<<1, 256, 0, stream>>>(query, Wk, Wv, Wo, rpe_bias, attn_scale_w,
                                           W1hi, W1lo, W3hi, W3lo, rpe, scl);

    qna_gemm_kernel<<<M1 / 16, 256, 0, stream>>>(x, W1hi, W1lo, 16, vbuf, cbuf);

    qna_conv_kernel<<<M3, 256, 0, stream>>>(cbuf, vbuf, rpe, scl, mid);

    qna_gemm_kernel<<<M3 / 16, 256, 0, stream>>>(mid, W3hi, W3lo, 16, (float*)d_out, nullptr);
}
